// NTXentLoss_57612691308847
// MI455X (gfx1250) — compile-verified, matrix operand order checked by reading, not run on hardware
//
#include <hip/hip_runtime.h>
#include <hip/hip_bf16.h>
#include <math.h>

// NT-Xent loss, B=4096, D=512, N=8192, TEMP=0.5
#define BB 4096
#define DD 512
#define NN 8192
#define INV_TEMP 2.0f
#define EPS_F 1e-8f

#define WAVES 4            // waves per block in the GEMM kernel (64 rows/block)
#define LDS_STRIDE 520     // 512 halves + 8 pad -> conflict-free ds_load_b128

typedef _Float16 v16h __attribute__((ext_vector_type(16)));
typedef _Float16 v8h  __attribute__((ext_vector_type(8)));
typedef _Float16 v4h  __attribute__((ext_vector_type(4)));
typedef float    v8f  __attribute__((ext_vector_type(8)));

// ---------------------------------------------------------------------------
// Kernel 1: row-normalize reps = concat([zjs, zis]) into f16 matrix zh[N][D].
// One wave (32 lanes) per row; 16 elements per lane.
// ---------------------------------------------------------------------------
__global__ __launch_bounds__(256)
void ntx_normalize(const float* __restrict__ zis,
                   const float* __restrict__ zjs,
                   _Float16* __restrict__ zh)
{
    const int wave = threadIdx.x >> 5;            // 8 waves / block
    const int lane = threadIdx.x & 31;
    const int row  = blockIdx.x * 8 + wave;       // 0..8191
    const float* __restrict__ src = (row < BB)
        ? (zjs + (size_t)row * DD)
        : (zis + (size_t)(row - BB) * DD);

    float4 v[4];
    float ss = 0.f;
#pragma unroll
    for (int i = 0; i < 4; ++i) {
        v[i] = ((const float4*)src)[lane + 32 * i];
        ss += v[i].x * v[i].x + v[i].y * v[i].y + v[i].z * v[i].z + v[i].w * v[i].w;
    }
#pragma unroll
    for (int m = 16; m >= 1; m >>= 1) ss += __shfl_xor(ss, m, 32);

    const float inv = 1.0f / fmaxf(sqrtf(ss), EPS_F);

    _Float16* __restrict__ dst = zh + (size_t)row * DD;
#pragma unroll
    for (int i = 0; i < 4; ++i) {
        v4h h;
        h[0] = (_Float16)(v[i].x * inv);
        h[1] = (_Float16)(v[i].y * inv);
        h[2] = (_Float16)(v[i].z * inv);
        h[3] = (_Float16)(v[i].w * inv);
        ((v4h*)dst)[lane + 32 * i] = h;
    }
}

// ---------------------------------------------------------------------------
// Kernel 2: fused sim = (zh * zh^T) * 2  +  per-row logsumexp (diag masked)
//           + positive-logit capture. Accumulates sum of per_row into *acc.
// Block: 128 threads = 4 waves, each wave owns one 16-row M tile (A resident
// in VGPRs), block iterates over all 512 column tiles with double-buffered
// LDS staging of the shared 16-row B tile.
// ---------------------------------------------------------------------------
__global__ __launch_bounds__(128)
void ntx_simloss(const _Float16* __restrict__ zh, float* __restrict__ acc)
{
    __shared__ _Float16 ldsB[2][16 * LDS_STRIDE];   // 2 x 16.25 KB

    const int tid  = threadIdx.x;
    const int lane = tid & 31;
    const int wave = tid >> 5;
    const int r16  = lane & 15;     // row/col index within a 16-tile
    const int hi   = lane >> 4;     // selects K sub-pattern per ISA A/B layout

    const int mbase = (blockIdx.x * WAVES + wave) * 16;

    // ---- Load resident A fragments: 16 rows x K=512 (16 fragments of 16x32).
    // ISA 16-bit A layout: lane<16 holds K {0..7,16..23}; lane>=16 {8..15,24..31}.
    v16h a[16];
    {
        const _Float16* rowp = zh + (size_t)(mbase + r16) * DD + hi * 8;
#pragma unroll
        for (int kt = 0; kt < 16; ++kt) {
            v8h lo = *(const v8h*)(rowp + kt * 32);
            v8h hh = *(const v8h*)(rowp + kt * 32 + 16);
#pragma unroll
            for (int e = 0; e < 8; ++e) { a[kt][e] = lo[e]; a[kt][8 + e] = hh[e]; }
        }
    }

    float se[8];   // running sum of exp(sim) per owned row (diag excluded)
    float pv[8];   // positive logit per owned row
#pragma unroll
    for (int j = 0; j < 8; ++j) { se[j] = 0.f; pv[j] = 0.f; }

    // Cooperative fill of one 16-row B tile into LDS (128 thr x 8 chunks of 16B)
    auto fill = [&](int nt, int buf) {
#pragma unroll
        for (int i = 0; i < 8; ++i) {
            const int cid = tid + 128 * i;        // 0..1023
            const int row = cid >> 6;             // 64 x 16B chunks per row
            const int col = (cid & 63) * 8;       // half index within row
            v8h g = *(const v8h*)(zh + (size_t)(nt * 16 + row) * DD + col);
            *(v8h*)(&ldsB[buf][row * LDS_STRIDE + col]) = g;
        }
    };

    const int NT = NN / 16;   // 512 column tiles
    fill(0, 0);
    int cur = 0;

    for (int nt = 0; nt < NT; ++nt) {
        __syncthreads();                       // buf[cur] ready; prior reads done
        if (nt + 1 < NT) fill(nt + 1, cur ^ 1);
        if (nt + 2 < NT) {                     // L2 prefetch two tiles ahead
            __builtin_prefetch((const char*)(zh + (size_t)((nt + 2) * 16 + r16) * DD), 0, 1);
        }

        // ---- 16 WMMAs over K=512 against the staged B tile.
        // B = A^T for z*z^T: lane = column n (= row n of zh), K-serial per lane,
        // same sub-pattern as the A fragment.
        const _Float16* bp = &ldsB[cur][r16 * LDS_STRIDE + hi * 8];
        v8f c = {};
#pragma unroll
        for (int kt = 0; kt < 16; ++kt) {
            v8h lo = *(const v8h*)(bp + kt * 32);
            v8h hh = *(const v8h*)(bp + kt * 32 + 16);
            v16h b;
#pragma unroll
            for (int e = 0; e < 8; ++e) { b[e] = lo[e]; b[8 + e] = hh[e]; }
            c = __builtin_amdgcn_wmma_f32_16x16x32_f16(
                    /*neg_a=*/false, a[kt], /*neg_b=*/false, b,
                    /*c_mod=*/(short)0, c, /*reuse_a=*/false, /*reuse_b=*/false);
        }

        // ---- Epilogue: C layout is VGPR j -> M = j + 8*hi, N = r16.
        const int gn = nt * 16 + r16;
#pragma unroll
        for (int j = 0; j < 8; ++j) {
            const int gm  = mbase + j + 8 * hi;
            const float s = c[j] * INV_TEMP;                   // sim value
            const float e = __expf(s);                        // |s| <= 4, safe
            se[j] += (gn == gm) ? 0.f : e;                    // mask diagonal
            if (gn == ((gm + BB) & (NN - 1))) pv[j] = s;      // positive pair
        }
        cur ^= 1;
    }

    // ---- Reduce across the 16 lanes that share each row (wave32 halves).
#pragma unroll
    for (int j = 0; j < 8; ++j) {
#pragma unroll
        for (int m = 8; m >= 1; m >>= 1) {
            se[j] += __shfl_xor(se[j], m, 32);
            pv[j] += __shfl_xor(pv[j], m, 32);
        }
    }
    if (r16 == 0) {   // lanes 0 and 16: rows mbase..+7 and mbase+8..+15
        float s = 0.f;
#pragma unroll
        for (int j = 0; j < 8; ++j) s += __logf(se[j]) - pv[j];
        atomicAdd(acc, s);
    }
}

// ---------------------------------------------------------------------------
__global__ void ntx_zero(float* acc) {
    if (threadIdx.x == 0) acc[0] = 0.f;
}

__global__ void ntx_finalize(const float* __restrict__ acc, float* __restrict__ out) {
    if (threadIdx.x == 0) out[0] = acc[0] * (1.0f / (float)NN);
}

// ---------------------------------------------------------------------------
extern "C" void kernel_launch(void* const* d_in, const int* in_sizes, int n_in,
                              void* d_out, int out_size, void* d_ws, size_t ws_size,
                              hipStream_t stream) {
    (void)in_sizes; (void)n_in; (void)out_size; (void)ws_size;
    const float* zis = (const float*)d_in[0];
    const float* zjs = (const float*)d_in[1];
    float* out = (float*)d_out;

    _Float16* zh = (_Float16*)d_ws;                                  // 8 MB f16
    float* acc   = (float*)((char*)d_ws + (size_t)NN * DD * sizeof(_Float16));

    ntx_normalize<<<NN / 8, 256, 0, stream>>>(zis, zjs, zh);
    ntx_zero<<<1, 32, 0, stream>>>(acc);
    ntx_simloss<<<NN / (16 * WAVES), 128, 0, stream>>>(zh, acc);
    ntx_finalize<<<1, 32, 0, stream>>>(acc, out);
}
